// ProGen_4887672783532
// MI455X (gfx1250) — matrix operand order repaired and checked by reading, not compiled, not run on hardware
//
#include <hip/hip_runtime.h>

#define VOCAB 32
#define DIM   256
#define SEQQ  8192
#define DEPTH 6
#define WSZ   512
#define HID   1024
#define BATCH 2
#define NTOK  (BATCH * SEQQ)

typedef __attribute__((ext_vector_type(16))) _Float16 v16h;
typedef __attribute__((ext_vector_type(8)))  _Float16 v8h;
typedef __attribute__((ext_vector_type(8)))  float    v8f;

__device__ __forceinline__ v8f vzero8() {
  v8f z;
#pragma unroll
  for (int i = 0; i < 8; ++i) z[i] = 0.0f;
  return z;
}

__device__ __forceinline__ v8f wmma_f16(v16h a, v16h b, v8f c) {
  // D = A(16x32 f16) * B(32x16 f16) + C(16x16 f32)
  return __builtin_amdgcn_wmma_f32_16x16x32_f16(
      /*neg_a=*/false, a, /*neg_b=*/false, b,
      /*c_mod=*/(short)0, c, /*reuse_a=*/false, /*reuse_b=*/false);
}

// A fragment (16x32, f16), source row-major: elem(m,k) = base[m*ld + k]
// Layout: lanes 0-15 row M=lane hold K {0..15}+k0,
//         lanes 16-31 row M=lane-16 hold K {16..31}+k0.
__device__ __forceinline__ v16h frag_a(const _Float16* base, int ld, int m0, int k0) {
  int lane = threadIdx.x & 31;
  const _Float16* p = base + (m0 + (lane & 15)) * ld + k0 + ((lane >> 4) << 4);
  v8h lo = *(const v8h*)p;
  v8h hi = *(const v8h*)(p + 8);
  return __builtin_shufflevector(lo, hi, 0,1,2,3,4,5,6,7,8,9,10,11,12,13,14,15);
}

// B fragment (32x16, f16) from column-major/transposed storage:
// elem(k,n) = base[n*ld + k]. Lane holds column N = n0+(lane&15), a contiguous
// 16-element K chunk at k0+(lane>>4)*16 -> two 128-bit loads (LDS or global).
__device__ __forceinline__ v16h frag_bt(const _Float16* base, int ld, int k0, int n0) {
  int lane = threadIdx.x & 31;
  const _Float16* p = base + (size_t)(n0 + (lane & 15)) * ld + k0 + ((lane >> 4) << 4);
  v8h lo = *(const v8h*)p;
  v8h hi = *(const v8h*)(p + 8);
  return __builtin_shufflevector(lo, hi, 0,1,2,3,4,5,6,7,8,9,10,11,12,13,14,15);
}

// ------------------------------------------------ weight f32 -> f16 transpose
// src: L layers of row-major [K][N] f32. dst: L layers of [N][K] f16.
__global__ __launch_bounds__(256) void cvt_tr_kernel(const float* __restrict__ src,
                                                     _Float16* __restrict__ dst,
                                                     int K, int N) {
  size_t i = (size_t)blockIdx.x * 256 + threadIdx.x;
  size_t perL = (size_t)K * N;           // power of two here
  size_t l = i / perL, r = i % perL;
  int k = (int)(r / N), n = (int)(r % N);
  dst[l * perL + (size_t)n * K + k] = (_Float16)src[i];
}

// ---------------------------------------------------------------- embedding
__global__ __launch_bounds__(256) void embed_kernel(const int* __restrict__ tokens,
                                                    const float* __restrict__ emb,
                                                    float* __restrict__ x) {
  size_t i = (size_t)blockIdx.x * 256 + threadIdx.x;
  int t = (int)(i >> 8);           // i / DIM
  int f = (int)(i & (DIM - 1));
  x[i] = emb[(size_t)tokens[t] * DIM + f];
}

// ---------------------------------------------------------------- attention
// One WG = 64 queries of one window of one batch. Streaming softmax over
// 64-key chunks spanning [prev window, current window] (1024 keys).
__global__ __launch_bounds__(256) void attn_kernel(const float* __restrict__ x,
                                                   float* __restrict__ y) {
  __shared__ _Float16 Qs[64 * DIM];   // rotary(q), row-major
  __shared__ _Float16 Ks[64 * DIM];   // rotary(k) chunk, row-major
  __shared__ _Float16 Vt[DIM * 64];   // v chunk, transposed [feat][key]
  __shared__ float    Ss[64 * 64];    // raw scores
  __shared__ _Float16 Ps[64 * 64];    // exp probabilities
  __shared__ float    sfac[64];
  __shared__ float    linv[64];

  const int tid  = threadIdx.x;
  const int lane = tid & 31;
  const int wv   = tid >> 5;
  const int g    = blockIdx.x;
  const int b    = g >> 7;            // 128 WGs per batch row
  const int w    = (g >> 3) & 15;
  const int qb   = g & 7;
  const int q0   = w * WSZ + qb * 64;
  const float* xb = x + (size_t)b * SEQQ * DIM;
  float*       yb = y + (size_t)b * SEQQ * DIM;

  const float KF = 0.103810253f;      // log2(10000)/128

  // Stage rotary(Q) into LDS.
  for (int idx = tid; idx < 64 * 128; idx += 256) {
    int r = idx >> 7, j = idx & 127;
    int pos = q0 + r;
    float inv = exp2f(-(float)j * KF);
    float ang = (float)pos * inv;
    float sn, cs; __sincosf(ang, &sn, &cs);
    float x1 = xb[(size_t)pos * DIM + j];
    float x2 = xb[(size_t)pos * DIM + j + 128];
    Qs[r * DIM + j]       = (_Float16)(x1 * cs - x2 * sn);
    Qs[r * DIM + j + 128] = (_Float16)(x2 * cs + x1 * sn);
  }
  __syncthreads();

  const int rtS = wv & 3;             // this wave's 16-row tile
  v16h qf[8];
#pragma unroll
  for (int k = 0; k < 8; ++k) qf[k] = frag_a(Qs, DIM, rtS * 16, k * 32);

  v8f o[8];                           // 16 rows x 128 cols of O per wave
#pragma unroll
  for (int j = 0; j < 8; ++j) o[j] = vzero8();
  float mrow = -1e30f, lrow = 0.0f;   // softmax state for row `tid` (tid<64)

  const int kstart = (w - 1) * WSZ;
  for (int kc = 0; kc < 16; ++kc) {
    int kp = kstart + kc * 64;
    if (kp + 63 < 0) continue;            // fully before sequence
    if (kp + 63 < q0 - WSZ) continue;     // fully outside exact window
    if (kp > q0 + 63) continue;           // fully in the future
    __syncthreads();

    // Stage rotary(K) row-major and V transposed.
    for (int idx = tid; idx < 64 * 128; idx += 256) {
      int r = idx >> 7, j = idx & 127;
      int pos = kp + r;
      float x1 = 0.0f, x2 = 0.0f;
      if (pos >= 0) {
        x1 = xb[(size_t)pos * DIM + j];
        x2 = xb[(size_t)pos * DIM + j + 128];
      }
      float inv = exp2f(-(float)j * KF);
      float ang = (float)(pos < 0 ? 0 : pos) * inv;
      float sn, cs; __sincosf(ang, &sn, &cs);
      Ks[r * DIM + j]        = (_Float16)(x1 * cs - x2 * sn);
      Ks[r * DIM + j + 128]  = (_Float16)(x2 * cs + x1 * sn);
      Vt[j * 64 + r]         = (_Float16)x1;
      Vt[(j + 128) * 64 + r] = (_Float16)x2;
    }
    __syncthreads();

    // S = Q K^T  (16 tiles, 2 per wave)
#pragma unroll
    for (int t = 0; t < 2; ++t) {
      int ct = ((wv >> 2) << 1) + t;
      v8f acc = vzero8();
#pragma unroll
      for (int k = 0; k < 8; ++k)
        acc = wmma_f16(qf[k], frag_bt(Ks, DIM, k * 32, ct * 16), acc);
#pragma unroll
      for (int i = 0; i < 8; ++i) {
        int row = rtS * 16 + i + ((lane >> 4) << 3);
        int col = ct * 16 + (lane & 15);
        Ss[row * 64 + col] = acc[i];
      }
    }
    __syncthreads();

    // Streaming softmax: thread tid < 64 owns row tid.
    if (tid < 64) {
      int tq = q0 + tid;
      float mc = -1e30f;
      for (int c = 0; c < 64; ++c) {
        int tk = kp + c;
        bool ok = (tk >= 0) && (tq >= tk) && (tq <= tk + WSZ);
        float sv = Ss[tid * 64 + c] * 0.0625f;   // d^-0.5
        if (ok && sv > mc) mc = sv;
      }
      float mnew = fmaxf(mrow, mc);
      float fac  = __expf(mrow - mnew);
      float ssum = 0.0f;
      for (int c = 0; c < 64; ++c) {
        int tk = kp + c;
        bool ok = (tk >= 0) && (tq >= tk) && (tq <= tk + WSZ);
        float sv = Ss[tid * 64 + c] * 0.0625f;
        float p  = ok ? __expf(sv - mnew) : 0.0f;
        Ps[tid * 64 + c] = (_Float16)p;
        ssum += p;
      }
      lrow = lrow * fac + ssum;
      mrow = mnew;
      sfac[tid] = fac;
    }
    __syncthreads();

    // O = O*fac + P V  (8 col tiles per wave, K=64 keys => 2 WMMA steps)
    v16h pf0 = frag_a(Ps, 64, rtS * 16, 0);
    v16h pf1 = frag_a(Ps, 64, rtS * 16, 32);
    float fct[8];
#pragma unroll
    for (int i = 0; i < 8; ++i) fct[i] = sfac[rtS * 16 + i + ((lane >> 4) << 3)];
    int cbase = (wv >> 2) * 128;
#pragma unroll
    for (int j = 0; j < 8; ++j) {
#pragma unroll
      for (int i = 0; i < 8; ++i) o[j][i] *= fct[i];
      o[j] = wmma_f16(pf0, frag_bt(Vt, 64, 0,  cbase + j * 16), o[j]);
      o[j] = wmma_f16(pf1, frag_bt(Vt, 64, 32, cbase + j * 16), o[j]);
    }
  }

  if (tid < 64) linv[tid] = 1.0f / lrow;
  __syncthreads();

  float fin[8];
#pragma unroll
  for (int i = 0; i < 8; ++i) fin[i] = linv[rtS * 16 + i + ((lane >> 4) << 3)];
  int cbase = (wv >> 2) * 128;
#pragma unroll
  for (int j = 0; j < 8; ++j) {
#pragma unroll
    for (int i = 0; i < 8; ++i) {
      int row = rtS * 16 + i + ((lane >> 4) << 3);
      int col = cbase + j * 16 + (lane & 15);
      size_t gi = (size_t)(q0 + row) * DIM + col;
      yb[gi] = xb[gi] + o[j][i] * fin[i];   // residual
    }
  }
}

// ---------------------------------------------------------------- fused LN+FFN
// One WG = 32 token rows. x_out = y + W2 @ gelu(W1 @ LN(y) + b1) + b2
// Weights are pre-converted f16, column-major [N][K].
__global__ __launch_bounds__(256) void ffn_kernel(const float* __restrict__ y,
                                                  const _Float16* __restrict__ w1t,
                                                  const float* __restrict__ b1,
                                                  const _Float16* __restrict__ w2t,
                                                  const float* __restrict__ b2,
                                                  const float* __restrict__ lns,
                                                  const float* __restrict__ lnb,
                                                  float* __restrict__ xo) {
  __shared__ _Float16 As[32 * DIM];   // LN(y) tile, f16
  __shared__ _Float16 Hs[32 * HID];   // gelu hidden tile, f16
  __shared__ float    redS[32][8];
  __shared__ float    redQ[32][8];
  __shared__ float    rmean[32];
  __shared__ float    rrstd[32];

  const int tid  = threadIdx.x;
  const int lane = tid & 31;
  const int wv   = tid >> 5;
  const size_t row0 = (size_t)blockIdx.x * 32;

  // ---- LayerNorm, 8 threads per row ----
  {
    int r = tid >> 3, seg = tid & 7;
    const float* yr = y + (row0 + r) * DIM + seg * 32;
    float sum = 0.0f, sq = 0.0f;
#pragma unroll
    for (int j = 0; j < 32; ++j) { float v = yr[j]; sum += v; sq += v * v; }
    redS[r][seg] = sum; redQ[r][seg] = sq;
  }
  __syncthreads();
  if (tid < 32) {
    float sum = 0.0f, sq = 0.0f;
#pragma unroll
    for (int s = 0; s < 8; ++s) { sum += redS[tid][s]; sq += redQ[tid][s]; }
    float mean = sum * (1.0f / DIM);
    float var  = sq * (1.0f / DIM) - mean * mean;
    rmean[tid] = mean;
    rrstd[tid] = rsqrtf(var + 1e-5f);
  }
  __syncthreads();
  {
    int r = tid >> 3, seg = tid & 7;
    float mean = rmean[r], rs = rrstd[r];
    const float* yr = y + (row0 + r) * DIM + seg * 32;
#pragma unroll
    for (int j = 0; j < 32; ++j) {
      int c = seg * 32 + j;
      As[r * DIM + c] = (_Float16)((yr[j] - mean) * rs * lns[c] + lnb[c]);
    }
  }
  __syncthreads();

  // A fragments for both 16-row tiles (shared across all column tiles).
  v16h af0[8], af1[8];
#pragma unroll
  for (int k = 0; k < 8; ++k) {
    af0[k] = frag_a(As, DIM, 0,  k * 32);
    af1[k] = frag_a(As, DIM, 16, k * 32);
  }

  // GEMM1 + bias + exact GELU -> Hs. Each wave: 8 column tiles, both row tiles.
  for (int t = 0; t < 8; ++t) {
    int ct = wv * 8 + t;
    if (t + 1 < 8)
      __builtin_prefetch(w1t + (size_t)((ct + 1) * 16) * DIM, 0, 1);
    v8f a0 = vzero8(), a1 = vzero8();
#pragma unroll
    for (int k = 0; k < 8; ++k) {
      v16h bf = frag_bt(w1t, DIM, k * 32, ct * 16);
      a0 = wmma_f16(af0[k], bf, a0);
      a1 = wmma_f16(af1[k], bf, a1);
    }
#pragma unroll
    for (int i = 0; i < 8; ++i) {
      int r0 = i + ((lane >> 4) << 3);
      int col = ct * 16 + (lane & 15);
      float v0 = a0[i] + b1[col];
      float v1 = a1[i] + b1[col];
      v0 = 0.5f * v0 * (1.0f + erff(v0 * 0.70710678f));
      v1 = 0.5f * v1 * (1.0f + erff(v1 * 0.70710678f));
      Hs[r0 * HID + col]        = (_Float16)v0;
      Hs[(r0 + 16) * HID + col] = (_Float16)v1;
    }
  }
  __syncthreads();

  // GEMM2 + bias + residual. Each wave: 2 column tiles, both row tiles.
  for (int t = 0; t < 2; ++t) {
    int ct = wv * 2 + t;
    if (t + 1 < 2)
      __builtin_prefetch(w2t + (size_t)((ct + 1) * 16) * HID, 0, 1);
    v8f a0 = vzero8(), a1 = vzero8();
    for (int k = 0; k < 32; ++k) {
      v16h bf = frag_bt(w2t, HID, k * 32, ct * 16);
      a0 = wmma_f16(frag_a(Hs, HID, 0,  k * 32), bf, a0);
      a1 = wmma_f16(frag_a(Hs, HID, 16, k * 32), bf, a1);
    }
#pragma unroll
    for (int i = 0; i < 8; ++i) {
      int r0 = i + ((lane >> 4) << 3);
      int col = ct * 16 + (lane & 15);
      size_t g0 = (row0 + r0) * DIM + col;
      size_t g1 = (row0 + r0 + 16) * DIM + col;
      xo[g0] = y[g0] + a0[i] + b2[col];
      xo[g1] = y[g1] + a1[i] + b2[col];
    }
  }
}

// ---------------------------------------------------------------- final LN + head
__global__ __launch_bounds__(256) void head_kernel(const float* __restrict__ x,
                                                   const float* __restrict__ lns,
                                                   const float* __restrict__ lnb,
                                                   const _Float16* __restrict__ wot,
                                                   const float* __restrict__ bo,
                                                   float* __restrict__ out) {
  __shared__ _Float16 As[64 * DIM];
  __shared__ float    redS[64][4];
  __shared__ float    redQ[64][4];
  __shared__ float    rmean[64];
  __shared__ float    rrstd[64];

  const int tid  = threadIdx.x;
  const int lane = tid & 31;
  const int wv   = tid >> 5;
  const size_t row0 = (size_t)blockIdx.x * 64;

  // ---- LayerNorm, 4 threads per row ----
  {
    int r = tid >> 2, seg = tid & 3;
    const float* xr = x + (row0 + r) * DIM + seg * 64;
    float sum = 0.0f, sq = 0.0f;
#pragma unroll
    for (int j = 0; j < 64; ++j) { float v = xr[j]; sum += v; sq += v * v; }
    redS[r][seg] = sum; redQ[r][seg] = sq;
  }
  __syncthreads();
  if (tid < 64) {
    float sum = redS[tid][0] + redS[tid][1] + redS[tid][2] + redS[tid][3];
    float sq  = redQ[tid][0] + redQ[tid][1] + redQ[tid][2] + redQ[tid][3];
    float mean = sum * (1.0f / DIM);
    float var  = sq * (1.0f / DIM) - mean * mean;
    rmean[tid] = mean;
    rrstd[tid] = rsqrtf(var + 1e-5f);
  }
  __syncthreads();
  {
    int r = tid >> 2, seg = tid & 3;
    float mean = rmean[r], rs = rrstd[r];
    const float* xr = x + (row0 + r) * DIM + seg * 64;
#pragma unroll
    for (int j = 0; j < 64; ++j) {
      int c = seg * 64 + j;
      As[r * DIM + c] = (_Float16)((xr[j] - mean) * rs * lns[c] + lnb[c]);
    }
  }
  __syncthreads();

  int rt = wv & 3, ct = wv >> 2;      // 8 waves, 8 tiles (64 x 32 output)
  v8f acc = vzero8();
#pragma unroll
  for (int k = 0; k < 8; ++k)
    acc = wmma_f16(frag_a(As, DIM, rt * 16, k * 32),
                   frag_bt(wot, DIM, k * 32, ct * 16), acc);
#pragma unroll
  for (int i = 0; i < 8; ++i) {
    int row = rt * 16 + i + ((lane >> 4) << 3);
    int col = ct * 16 + (lane & 15);
    out[(row0 + row) * VOCAB + col] = acc[i] + bo[col];
  }
}

// ---------------------------------------------------------------- launcher
extern "C" void kernel_launch(void* const* d_in, const int* in_sizes, int n_in,
                              void* d_out, int out_size, void* d_ws, size_t ws_size,
                              hipStream_t stream) {
  const int*   tokens  = (const int*)d_in[0];
  const float* embw    = (const float*)d_in[1];
  const float* ff_ln_s = (const float*)d_in[2];
  const float* ff_ln_b = (const float*)d_in[3];
  const float* w1      = (const float*)d_in[4];
  const float* b1      = (const float*)d_in[5];
  const float* w2      = (const float*)d_in[6];
  const float* b2      = (const float*)d_in[7];
  const float* lnf_s   = (const float*)d_in[8];
  const float* lnf_b   = (const float*)d_in[9];
  const float* w_out   = (const float*)d_in[10];
  const float* b_out   = (const float*)d_in[11];
  float* out = (float*)d_out;

  float* xb = (float*)d_ws;                     // residual stream (ping)
  float* yb = xb + (size_t)NTOK * DIM;          // residual stream (pong)
  _Float16* w1h = (_Float16*)(yb + (size_t)NTOK * DIM);
  _Float16* w2h = w1h + (size_t)DEPTH * DIM * HID;
  _Float16* woh = w2h + (size_t)DEPTH * HID * DIM;

  // One-time (per launch) weight convert+transpose to f16 [N][K].
  cvt_tr_kernel<<<(DEPTH * DIM * HID) / 256, 256, 0, stream>>>(w1, w1h, DIM, HID);
  cvt_tr_kernel<<<(DEPTH * HID * DIM) / 256, 256, 0, stream>>>(w2, w2h, HID, DIM);
  cvt_tr_kernel<<<(DIM * VOCAB) / 256, 256, 0, stream>>>(w_out, woh, DIM, VOCAB);

  embed_kernel<<<(NTOK * DIM) / 256, 256, 0, stream>>>(tokens, embw, xb);

  for (int l = 0; l < DEPTH; ++l) {
    attn_kernel<<<BATCH * (SEQQ / WSZ) * (WSZ / 64), 256, 0, stream>>>(xb, yb);
    ffn_kernel<<<NTOK / 32, 256, 0, stream>>>(
        yb,
        w1h + (size_t)l * DIM * HID, b1 + (size_t)l * HID,
        w2h + (size_t)l * HID * DIM, b2 + (size_t)l * DIM,
        ff_ln_s + (size_t)l * DIM,   ff_ln_b + (size_t)l * DIM,
        xb);
  }

  head_kernel<<<NTOK / 64, 256, 0, stream>>>(xb, lnf_s, lnf_b, woh, b_out, out);
}
